// HierarchicalFilter_14250701488167
// MI455X (gfx1250) — matrix operand order repaired and checked by reading, not compiled, not run on hardware
//
#include <hip/hip_runtime.h>
#include <hip/hip_bf16.h>
#include <math.h>
#include <stdint.h>

// ---------------- CDNA5 (gfx1250, wave32) types ----------------
typedef __bf16 bf16_t;
typedef __attribute__((ext_vector_type(16))) bf16_t v16bf;
typedef __attribute__((ext_vector_type(8)))  float  v8f;
typedef __attribute__((ext_vector_type(4)))  int    v4i;
typedef __attribute__((ext_vector_type(2)))  int    v2i;

#define T_TOK 4096
#define HD    512               // H == D == Dv == 512
#define BATCH 8
#define MTOT  (BATCH * T_TOK)   // 32768 rows of flattened [B*T, H]
#define NKT   (HD / 32)         // 16 K-tiles

#define AS1 __attribute__((address_space(1)))
#define AS3 __attribute__((address_space(3)))

#if __has_builtin(__builtin_amdgcn_global_load_async_to_lds_b128) && \
    __has_builtin(__builtin_amdgcn_global_load_async_to_lds_b64)
#define HAVE_ASYNC_LDS 1
#else
#define HAVE_ASYNC_LDS 0
#endif

// typed address-space casts for the async-to-LDS builtins
static __device__ __forceinline__ AS1 v4i* gbl4(const void* p) {
  return (AS1 v4i*)(uintptr_t)p;
}
static __device__ __forceinline__ AS1 v2i* gbl2(const void* p) {
  return (AS1 v2i*)(uintptr_t)p;
}
static __device__ __forceinline__ AS3 v4i* lds4(void* p) {
  return (AS3 v4i*)(uint32_t)(uintptr_t)p;   // generic -> 32-bit LDS offset -> AS3
}
static __device__ __forceinline__ AS3 v2i* lds2(void* p) {
  return (AS3 v2i*)(uint32_t)(uintptr_t)p;
}

// f32 -> bf16 (RNE, used in cold prologue only)
static __device__ __forceinline__ unsigned short f2bf(float f) {
  unsigned u = __float_as_uint(f);
  unsigned r = 0x7FFFu + ((u >> 16) & 1u);
  return (unsigned short)((u + r) >> 16);
}
// hot-loop pack: two f32 -> packed bf16 pair, single v_perm_b32 (truncation;
// <=1 ulp of an 8-bit mantissa vs RNE, negligible vs bf16 quantization)
static __device__ __forceinline__ unsigned pack_bf16(float lo, float hi) {
  return __builtin_amdgcn_perm(__float_as_uint(hi), __float_as_uint(lo), 0x07060302u);
}

union FragAB {
  v16bf        v;
  unsigned int u[8];
};

// hash-based Gumbel noise (forward value of straight-through sample is the
// hard one-hot; JAX threefry cannot be matched bitwise in any case)
static __device__ __forceinline__ float gumbel(unsigned m, unsigned i, unsigned c) {
  unsigned x = m * 0x9E3779B9u ^ (i * 0x85EBCA6Bu + c * 0xC2B2AE35u + 0x27220A95u);
  x ^= x >> 16; x *= 0x7FEB352Du; x ^= x >> 15; x *= 0x846CA68Bu; x ^= x >> 16;
  float u = ((float)(x >> 8) + 0.5f) * (1.0f / 16777216.0f);
  return -__logf(-__logf(u));
}

// ---------------------------------------------------------------------------
// Prologue: W_embed [k][n] f32 -> Wt [n][k] bf16 (1 MB once per call).
// ---------------------------------------------------------------------------
__global__ __launch_bounds__(256) void conv_transpose_w(
    const float* __restrict__ W, unsigned short* __restrict__ Wt)
{
  const int id = blockIdx.x * 256 + threadIdx.x;   // 512*512 ids
  const int n = id >> 9, k = id & 511;
  Wt[(size_t)n * HD + k] = f2bf(W[(size_t)k * HD + n]);
}

// ---------------------------------------------------------------------------
// Kernel 1: h = (data @ W_embed + b_embed + PE(t)) * 8, fused with the
// per-token logit contraction  logits[m][j] += h[m,:] . Wx[:,j]  (atomics).
// 128x128 block tile, 8 waves of 32x64, v_wmma_f32_16x16x32_bf16.
// Double-buffered async-to-LDS staging (ASYNCcnt), copies shadowed by WMMA.
// ---------------------------------------------------------------------------
__global__ __launch_bounds__(256) void gemm_embed_pe(
    const float* __restrict__ data,        // [MTOT, 512] f32
    const unsigned short* __restrict__ Wt, // [512 n][512 k] bf16
    const float* __restrict__ bias,        // [512]
    const float* __restrict__ Wx,          // [512, 2]
    float* __restrict__ hout,              // [MTOT, 512]
    float* __restrict__ logits)            // [MTOT, 2], pre-zeroed
{
  __shared__ float          Af[2][128][36]; // m-major f32 (144B pitch, 16B aligned)
  __shared__ unsigned short Bs[2][128][36]; // n-major bf16 (72B pitch, 8B aligned)

  const int tid  = threadIdx.x;
  const int lane = tid & 31;            // wave32
  const int wid  = tid >> 5;            // 8 waves
  const int wm   = (wid & 3) * 32;
  const int wn   = (wid >> 2) * 64;
  const int hi   = lane >> 4;           // half-wave (ISA K split)
  const int ln16 = lane & 15;

  const int gm0 = blockIdx.y * 128;
  const int gn0 = blockIdx.x * 128;

  // stage one K-tile (32 wide) into LDS buffer `buf`
  auto issue_tile = [&](int kt, int buf) {
    const int k0 = kt * 32;
#if HAVE_ASYNC_LDS
    #pragma unroll
    for (int it = 0; it < 4; ++it) {
      const int idx = tid + it * 256;          // 1024 x 16B chunks (A)
      const int row = idx >> 3, part = (idx & 7) * 4;
      __builtin_amdgcn_global_load_async_to_lds_b128(
          gbl4(data + (size_t)(gm0 + row) * HD + k0 + part),
          lds4(&Af[buf][row][part]), 0, 0);
    }
    #pragma unroll
    for (int it = 0; it < 4; ++it) {
      const int idx = tid + it * 256;          // 1024 x 8B chunks (B)
      const int row = idx >> 3, part = (idx & 7) * 4;
      __builtin_amdgcn_global_load_async_to_lds_b64(
          gbl2(Wt + (size_t)(gn0 + row) * HD + k0 + part),
          lds2(&Bs[buf][row][part]), 0, 0);
    }
#else
    #pragma unroll
    for (int it = 0; it < 4; ++it) {
      const int idx = tid + it * 256;
      const int row = idx >> 3, part = (idx & 7) * 4;
      const float4 f = *(const float4*)(data + (size_t)(gm0 + row) * HD + k0 + part);
      *(float4*)&Af[buf][row][part] = f;
    }
    #pragma unroll
    for (int it = 0; it < 4; ++it) {
      const int idx = tid + it * 256;
      const int row = idx >> 3, part = (idx & 7) * 4;
      const uint2 w = *(const uint2*)(Wt + (size_t)(gn0 + row) * HD + k0 + part);
      *(uint2*)&Bs[buf][row][part] = w;
    }
#endif
  };

  v8f zero = {0.f,0.f,0.f,0.f,0.f,0.f,0.f,0.f};
  v8f acc[2][4];
  #pragma unroll
  for (int i = 0; i < 2; ++i)
    #pragma unroll
    for (int j = 0; j < 4; ++j) acc[i][j] = zero;

  issue_tile(0, 0);

  for (int kt = 0; kt < NKT; ++kt) {
    const int buf = kt & 1;
#if HAVE_ASYNC_LDS
#if __has_builtin(__builtin_amdgcn_s_wait_asynccnt)
    __builtin_amdgcn_s_wait_asynccnt(0);
#else
    asm volatile("s_wait_asynccnt 0x0" ::: "memory");
#endif
#endif
    // tile kt landed in buf; all waves also done reading buf^1 (last iter)
    __syncthreads();
    if (kt + 1 < NKT) issue_tile(kt + 1, buf ^ 1);  // overlap copy with WMMA

    // fragments per ISA 16-bit A 16x32 / B 32x16 wave32 layouts
    FragAB a[2], b[4];
    #pragma unroll
    for (int im = 0; im < 2; ++im) {
      const float* rp = &Af[buf][wm + im * 16 + ln16][0];
      #pragma unroll
      for (int j = 0; j < 4; ++j) {
        const float2 f0 = *(const float2*)(rp + hi * 8 + 2 * j);        // K=hi*8+2j
        const float2 f1 = *(const float2*)(rp + 16 + hi * 8 + 2 * j);   // K=16+hi*8+2j
        a[im].u[j]     = pack_bf16(f0.x, f0.y);
        a[im].u[4 + j] = pack_bf16(f1.x, f1.y);
      }
    }
    #pragma unroll
    for (int in = 0; in < 4; ++in) {
      const unsigned int* rp = (const unsigned int*)&Bs[buf][wn + in * 16 + ln16][0];
      #pragma unroll
      for (int j = 0; j < 4; ++j) {
        b[in].u[j]     = rp[hi * 4 + j];                                 // K=hi*8+2j
        b[in].u[4 + j] = rp[8 + hi * 4 + j];                             // K=16+hi*8+2j
      }
    }
    #pragma unroll
    for (int im = 0; im < 2; ++im)
      #pragma unroll
      for (int in = 0; in < 4; ++in)
        acc[im][in] = __builtin_amdgcn_wmma_f32_16x16x32_bf16(
            false, a[im].v, false, b[in].v, (short)0, acc[im][in], false, false);
  }

  // epilogue: + bias + PE, *8; fused h.Wx partial contraction with
  // 16-lane butterfly reduce + one f32 atomic per (row-half, j)
  const float coef = -0.03597867f;  // -ln(10000)/256
  float be[4], freq[4], wx0[4], wx1[4];
  int   gnc[4], godd[4];
  #pragma unroll
  for (int in = 0; in < 4; ++in) {
    const int gn = gn0 + wn + in * 16 + ln16;
    gnc[in]  = gn;
    godd[in] = gn & 1;
    be[in]   = bias[gn];
    freq[in] = __expf((float)(gn >> 1) * coef);
    wx0[in]  = Wx[2 * gn + 0];
    wx1[in]  = Wx[2 * gn + 1];
  }
  #pragma unroll
  for (int im = 0; im < 2; ++im) {
    #pragma unroll
    for (int r = 0; r < 8; ++r) {
      const int gm = gm0 + wm + im * 16 + hi * 8 + r;  // C/D: vgpr r -> M=r / r+8
      const int t  = gm & (T_TOK - 1);
      float p0 = 0.f, p1 = 0.f;
      #pragma unroll
      for (int in = 0; in < 4; ++in) {
        const float ang = (float)t * freq[in];
        const float pe  = godd[in] ? __cosf(ang) : __sinf(ang);
        const float val = (acc[im][in][r] + be[in] + pe) * 8.0f;
        hout[(size_t)gm * HD + gnc[in]] = val;
        p0 = fmaf(val, wx0[in], p0);
        p1 = fmaf(val, wx1[in], p1);
      }
      #pragma unroll
      for (int off = 1; off < 16; off <<= 1) {   // stays within each 16-lane half
        p0 += __shfl_xor(p0, off, 32);
        p1 += __shfl_xor(p1, off, 32);
      }
      if (ln16 == 0) {                            // lanes 0 and 16 (distinct gm)
        atomicAdd(&logits[(size_t)gm * 2 + 0], p0);
        atomicAdd(&logits[(size_t)gm * 2 + 1], p1);
      }
    }
  }
}

// ---------------------------------------------------------------------------
// Kernel 2a: per-batch value bias  vb[b][i][j] = v_i[b,:] . Wv[:,j]
// ---------------------------------------------------------------------------
__global__ __launch_bounds__(64) void v_bias(
    const float* __restrict__ v0, const float* __restrict__ v1,
    const float* __restrict__ Wv, float* __restrict__ vb)
{
  const int b = blockIdx.x;
  const int i = threadIdx.x >> 5;
  const int lane = threadIdx.x & 31;
  const float* v = i ? v1 : v0;
  float s0 = 0.f, s1 = 0.f;
  for (int e = lane; e < HD; e += 32) {
    const float ve = v[b * HD + e];
    s0 = fmaf(ve, Wv[2 * e + 0], s0);
    s1 = fmaf(ve, Wv[2 * e + 1], s1);
  }
  #pragma unroll
  for (int off = 16; off; off >>= 1) {
    s0 += __shfl_xor(s0, off, 32);
    s1 += __shfl_xor(s1, off, 32);
  }
  if (lane == 0) { vb[b * 4 + i * 2 + 0] = s0; vb[b * 4 + i * 2 + 1] = s1; }
}

// ---------------------------------------------------------------------------
// Kernel 2b: gumbel-argmax mask, 1 thread per token (logits pre-contracted).
// ---------------------------------------------------------------------------
__global__ __launch_bounds__(256) void make_mask(
    const float* __restrict__ logits, const float* __restrict__ vb,
    const float* __restrict__ bfv, int* __restrict__ maskOut)
{
  const int m = blockIdx.x * 256 + threadIdx.x;
  const int b = m >> 12;
  const float a0 = logits[(size_t)m * 2 + 0] + bfv[0];
  const float a1 = logits[(size_t)m * 2 + 1] + bfv[1];
  const int f0 = (a0 + vb[b * 4 + 0] + gumbel(m, 0, 0)) >=
                 (a1 + vb[b * 4 + 1] + gumbel(m, 0, 1));
  const int f1 = (a0 + vb[b * 4 + 2] + gumbel(m, 1, 0)) >=
                 (a1 + vb[b * 4 + 3] + gumbel(m, 1, 1));
  maskOut[m] = f0 & f1;
}

// ---------------------------------------------------------------------------
// Kernel 3a: per-batch-row stable compaction indices via block-wide scan.
// ---------------------------------------------------------------------------
__global__ __launch_bounds__(1024) void row_scan(
    const int* __restrict__ mask, int* __restrict__ dest)
{
  __shared__ int sc[1024];
  const int tid  = threadIdx.x;
  const int base = blockIdx.x * T_TOK + tid * 4;
  const int m0 = mask[base + 0], m1 = mask[base + 1];
  const int m2 = mask[base + 2], m3 = mask[base + 3];
  const int s = m0 + m1 + m2 + m3;
  sc[tid] = s;
  __syncthreads();
  for (int off = 1; off < 1024; off <<= 1) {
    const int v = (tid >= off) ? sc[tid - off] : 0;
    __syncthreads();
    sc[tid] += v;
    __syncthreads();
  }
  int excl = sc[tid] - s;
  dest[base + 0] = m0 ? excl : -1; excl += m0;
  dest[base + 1] = m1 ? excl : -1; excl += m1;
  dest[base + 2] = m2 ? excl : -1; excl += m2;
  dest[base + 3] = m3 ? excl : -1;
}

// ---------------------------------------------------------------------------
// Kernel 3b/3c: zero fill + scatter selected rows to packed positions.
// ---------------------------------------------------------------------------
__global__ void zero_out(float4* __restrict__ o, int n4) {
  const int i = blockIdx.x * blockDim.x + threadIdx.x;
  if (i < n4) o[i] = make_float4(0.f, 0.f, 0.f, 0.f);
}

__global__ __launch_bounds__(256) void scatter_rows(
    const float* __restrict__ h, const int* __restrict__ dest,
    float* __restrict__ out)
{
  const int lane = threadIdx.x & 31;
  const int m = blockIdx.x * 8 + (threadIdx.x >> 5);
  const int d = dest[m];
  if (d < 0) return;
  const int b = m >> 12;
  const float4* src = (const float4*)(h + (size_t)m * HD);
  float4* dst = (float4*)(out + ((size_t)b * T_TOK + d) * HD);
  #pragma unroll
  for (int j = 0; j < 4; ++j) dst[lane + j * 32] = src[lane + j * 32];
}

// ---------------------------------------------------------------------------
extern "C" void kernel_launch(void* const* d_in, const int* in_sizes, int n_in,
                              void* d_out, int out_size, void* d_ws, size_t ws_size,
                              hipStream_t stream)
{
  const float* data = (const float*)d_in[0];
  const float* v0   = (const float*)d_in[1];
  const float* v1   = (const float*)d_in[2];
  const float* W    = (const float*)d_in[3];
  const float* be   = (const float*)d_in[4];
  const float* Wx   = (const float*)d_in[5];
  const float* Wv   = (const float*)d_in[6];
  const float* bfv  = (const float*)d_in[7];
  float* out = (float*)d_out;

  // workspace layout
  float*          hbuf   = (float*)d_ws;                      // 32768*512 f32
  unsigned short* Wt     = (unsigned short*)(hbuf + (size_t)MTOT * HD); // 512*512 bf16
  float*          logits = (float*)(Wt + (size_t)HD * HD);    // 32768*2 f32
  float*          vb     = logits + (size_t)MTOT * 2;         // 32 f32
  int*            maskA  = (int*)(vb + 32);                   // 32768
  int*            destA  = maskA + MTOT;                      // 32768

  const int nlog4 = MTOT * 2 / 4;
  zero_out<<<(nlog4 + 255) / 256, 256, 0, stream>>>((float4*)logits, nlog4);
  conv_transpose_w<<<(HD * HD) / 256, 256, 0, stream>>>(W, Wt);

  dim3 g1(HD / 128, MTOT / 128);                              // (4, 256)
  gemm_embed_pe<<<g1, 256, 0, stream>>>(data, Wt, be, Wx, hbuf, logits);

  v_bias<<<BATCH, 64, 0, stream>>>(v0, v1, Wv, vb);
  make_mask<<<MTOT / 256, 256, 0, stream>>>(logits, vb, bfv, maskA);
  row_scan<<<BATCH, 1024, 0, stream>>>(maskA, destA);

  const int n4 = MTOT * HD / 4;
  zero_out<<<(n4 + 255) / 256, 256, 0, stream>>>((float4*)out, n4);
  scatter_rows<<<MTOT / 8, 256, 0, stream>>>(hbuf, destA, out);
}
